// Decoder_TNTM_59648505807412
// MI455X (gfx1250) — matrix operand order repaired and checked by reading, not compiled, not run on hardware
//
#include <hip/hip_runtime.h>
#include <hip/hip_bf16.h>

// ---------------------------------------------------------------------------
// Low-rank MVN mixture log-prob on gfx1250.
// Heavy GEMM (E @ U_k, ~105 GFLOP fp32-equivalent) on bf16 WMMA with hi/lo
// split (3 bf16 GEMMs ~ fp32 accuracy at ~2.7x the f32-WMMA rate).
// U_k fragment blocks + raw E tile staged to LDS with the Tensor Data Mover
// (tensor_load_to_lds), double-buffered over k, tracked with TENSORcnt.
// ---------------------------------------------------------------------------

typedef __attribute__((ext_vector_type(16))) __bf16 v16bf;
typedef __attribute__((ext_vector_type(8)))  float  v8f;
typedef __attribute__((ext_vector_type(4)))  unsigned int v4u;
typedef __attribute__((ext_vector_type(8)))  int    v8i;
typedef __attribute__((ext_vector_type(4)))  int    v4i;

#define DDIM 256
#define RDIM 64
#define KCOMP 64
#define BDIM 64
#define TV 128            // V rows per workgroup in main kernel
#define LOG2PI 1.8378770664093453f

// ---- fragment layout helpers (B-matrix, 16x16x32 bf16, wave32) -------------
__device__ __forceinline__ int frag_index(int d, int j) {
    int ks   = d >> 5;
    int kkk  = d & 31;
    int lane = (j & 15) + ((kkk & 16) ? 16 : 0);
    int e    = kkk & 15;
    int nt   = j >> 4;
    return ((nt * 8 + ks) * 32 + lane) * 16 + e;
}

__device__ __forceinline__ void store_frag_hilo(__bf16* base, int idx, float v) {
    __bf16 h = (__bf16)v;
    base[idx]         = h;                       // hi block [0,16384)
    base[16384 + idx] = (__bf16)(v - (float)h);  // lo block [16384,32768)
}

// ---- Tensor Data Mover: 1-D tile of `tile8` 8-byte elements -> LDS ---------
// OOB reads (beyond tensor8 valid elements) return zero -> free edge padding.
__device__ __forceinline__ void tdm_load_to_lds(const void* gsrc, unsigned lds_off,
                                                unsigned tile8, unsigned tensor8) {
    unsigned long long ga = (unsigned long long)gsrc;
    v4u g0;
    g0[0] = 1u;                                   // count=1, user desc
    g0[1] = lds_off;                              // LDS byte address
    g0[2] = (unsigned)ga;                         // global_addr[31:0]
    g0[3] = (unsigned)(ga >> 32) | (2u << 30);    // global_addr[56:32] | type=2
    v8i g1;
    g1[0] = 0x30000;                              // wg_mask=0, data_size=3 (8B)
    g1[1] = (int)((tensor8 & 0xFFFFu) << 16);     // tensor_dim0[15:0]
    g1[2] = (int)((tensor8 >> 16) | (1u << 16));  // tensor_dim0[31:16] | tensor_dim1=1
    g1[3] = (int)(tile8 << 16);                   // tile_dim0
    g1[4] = 1;                                    // tile_dim1=1, tile_dim2=0
    g1[5] = (int)tensor8;                         // tensor_dim0_stride lo (unused, 1 row)
    g1[6] = 0;
    g1[7] = 0;
    v4i g2; g2[0] = 1; g2[1] = 1; g2[2] = 0; g2[3] = 0;   // tensor_dim2/3 = 1
    v4i g3; g3[0] = 0; g3[1] = (int)(1u << 16); g3[2] = 0; g3[3] = 0; // tensor_dim4=1
#if __clang_major__ >= 23
    v8i gx = {0, 0, 0, 0, 0, 0, 0, 0};
    __builtin_amdgcn_tensor_load_to_lds(g0, g1, g2, g3, gx, 0);
#else
    __builtin_amdgcn_tensor_load_to_lds(g0, g1, g2, g3, 0);
#endif
}

// ---------------------------------------------------------------------------
// Kernel 1: per-component precompute.  One WG (256 thr) per component k.
// ---------------------------------------------------------------------------
__global__ void __launch_bounds__(256)
precompute_kernel(const float* __restrict__ mus,
                  const float* __restrict__ L,
                  const float* __restrict__ logd,
                  __bf16* __restrict__ Ufrag,     // [K][32768] bf16 (hi|lo)
                  __bf16* __restrict__ DinvFrag,  // [32768] bf16 (hi|lo)
                  __bf16* __restrict__ DmuFrag,   // [32768] bf16 (hi|lo)
                  float*  __restrict__ u0,        // [K][64]
                  float*  __restrict__ cAll)      // [K]
{
    __shared__ float sWd[DDIM * RDIM];   // 64 KB
    __shared__ float sC[RDIM * RDIM];    // 16 KB (C -> Lc in place)
    __shared__ float sLinv[RDIM * RDIM]; // 16 KB
    __shared__ float sDexp[DDIM], sDinv[DDIM], sMu[DDIM];
    __shared__ float sRed[256];
    __shared__ float smWd[RDIM];

    const int t = threadIdx.x;
    const int k = blockIdx.x;

    float ld   = logd[k * DDIM + t];
    float dinv = expf(-ld);
    float mu   = mus[k * DDIM + t];
    sDexp[t] = expf(ld);
    sDinv[t] = dinv;
    sMu[t]   = mu;

    // reductions: sum(log_diag), mu^T D^-1 mu
    sRed[t] = ld;
    __syncthreads();
    for (int st = 128; st > 0; st >>= 1) {
        if (t < st) sRed[t] += sRed[t + st];
        __syncthreads();
    }
    float sumlog = sRed[0];
    __syncthreads();
    sRed[t] = dinv * mu * mu;
    __syncthreads();
    for (int st = 128; st > 0; st >>= 1) {
        if (t < st) sRed[t] += sRed[t + st];
        __syncthreads();
    }
    float muDmu = sRed[0];
    __syncthreads();

    // Wd = D^-1 W   (coalesced over L)
    for (int o = t; o < DDIM * RDIM; o += 256) {
        int d = o >> 6;
        sWd[o] = L[(size_t)k * DDIM * RDIM + o] * sDinv[d];
    }
    __syncthreads();

    // C = I + Wd^T D Wd  (all from LDS)
    for (int o = t; o < RDIM * RDIM; o += 256) {
        int a = o >> 6, b = o & 63;
        float s = (a == b) ? 1.0f : 0.0f;
        for (int d = 0; d < DDIM; ++d)
            s += sWd[d * RDIM + a] * sWd[d * RDIM + b] * sDexp[d];
        sC[o] = s;
    }
    __syncthreads();

    // In-place right-looking Cholesky (lower triangle)
    for (int j = 0; j < RDIM; ++j) {
        if (t == 0) sC[j * RDIM + j] = sqrtf(sC[j * RDIM + j]);
        __syncthreads();
        if (t > j && t < RDIM) sC[t * RDIM + j] /= sC[j * RDIM + j];
        __syncthreads();
        if (t > j && t < RDIM) {
            float ltj = sC[t * RDIM + j];
            for (int p = j + 1; p <= t; ++p)
                sC[t * RDIM + p] -= ltj * sC[p * RDIM + j];
        }
        __syncthreads();
    }

    if (t == 0) {
        float acc = 0.0f;
        for (int j = 0; j < RDIM; ++j) acc += logf(sC[j * RDIM + j]);
        float logdet = sumlog + 2.0f * acc;
        cAll[k] = -0.5f * ((float)DDIM * LOG2PI + logdet + muDmu);
    }

    // Linv = Lc^-1 (forward substitution; thread t owns column t)
    for (int o = t; o < RDIM * RDIM; o += 256) sLinv[o] = 0.0f;
    __syncthreads();
    if (t < RDIM) {
        for (int i = t; i < RDIM; ++i) {
            float s = (i == t) ? 1.0f : 0.0f;
            for (int p = t; p < i; ++p) s -= sC[i * RDIM + p] * sLinv[p * RDIM + t];
            sLinv[i * RDIM + t] = s / sC[i * RDIM + i];
        }
    }
    __syncthreads();

    // mWd[p] = sum_d Wd[d][p] * mu[d];  u0[j] = sum_p Linv[j][p] * mWd[p]
    if (t < RDIM) {
        float s = 0.0f;
        for (int d = 0; d < DDIM; ++d) s += sWd[d * RDIM + t] * sMu[d];
        smWd[t] = s;
    }
    __syncthreads();
    if (t < RDIM) {
        float s = 0.0f;
        for (int p = 0; p < RDIM; ++p) s += sLinv[t * RDIM + p] * smWd[p];
        u0[k * RDIM + t] = s;
    }

    // U = Wd @ Linv^T -> hi/lo B-fragments
    __bf16* uf = Ufrag + (size_t)k * 32768;
    for (int o = t; o < DDIM * RDIM; o += 256) {
        int d = o >> 6, j = o & 63;
        float s = 0.0f;
        for (int p = 0; p < RDIM; ++p) s += sWd[d * RDIM + p] * sLinv[j * RDIM + p];
        store_frag_hilo(uf, frag_index(d, j), s);
    }

    // This block owns column j=k of the Dinv / Dmu B-matrices
    {
        int idx = frag_index(t, k);
        store_frag_hilo(DinvFrag, idx, sDinv[t]);
        store_frag_hilo(DmuFrag,  idx, sDinv[t] * sMu[t]);
    }
}

// ---------------------------------------------------------------------------
// Kernel 2: softmax of theta_hat rows -> T[b][k]
// ---------------------------------------------------------------------------
__global__ void softmax_kernel(const float* __restrict__ theta,
                               float* __restrict__ T)
{
    const int b = blockIdx.x, t = threadIdx.x;
    __shared__ float s[KCOMP];
    float x = theta[b * KCOMP + t];
    s[t] = x;
    __syncthreads();
    for (int st = 32; st > 0; st >>= 1) {
        if (t < st) s[t] = fmaxf(s[t], s[t + st]);
        __syncthreads();
    }
    float m = s[0];
    __syncthreads();
    float e = expf(x - m);
    s[t] = e;
    __syncthreads();
    for (int st = 32; st > 0; st >>= 1) {
        if (t < st) s[t] += s[t + st];
        __syncthreads();
    }
    T[b * KCOMP + t] = e / s[0];
}

// ---------------------------------------------------------------------------
// Kernel 3: main GEMM kernel.  One WG = 8 waves, TV=128 V-rows, all K comps.
// LDS map (196608 B):
//   phase 1: [0,131072)   raw E tile f32 [128][256]   (via TDM, zero-padded)
//   phase 2: [0,65536)    B-frag buffer 0 (hi|lo)     (TDM double buffer)
//            [65536,131072)  B-frag buffer 1 (hi|lo)
//   always:  [131072,163840) sS2[k][row] | [163840,196608) sQ[k][row]
// ---------------------------------------------------------------------------
__global__ void __launch_bounds__(256)
main_kernel(const float* __restrict__ emb,
            const __bf16* __restrict__ Ufrag,
            const __bf16* __restrict__ DinvFrag,
            const __bf16* __restrict__ DmuFrag,
            const float* __restrict__ u0,
            const float* __restrict__ cAll,
            float* __restrict__ lb,
            int V)
{
    __shared__ __align__(16) unsigned char smem[196608];
    __shared__ float sCk[KCOMP];

    float* sRaw = (float*)smem;
    float* sS2  = (float*)(smem + 131072);
    float* sQ   = (float*)(smem + 163840);

    const int tid  = threadIdx.x;
    const int wave = tid >> 5;
    const int lane = tid & 31;
    const int lr   = lane & 15;
    const int hi8  = (lane & 16) >> 1;   // 0 or 8
    const int vbase = blockIdx.x * TV;
    const unsigned ldsbase = (unsigned)(unsigned long long)(uintptr_t)smem;

    // ---- TDM: raw E tile -> LDS, rows past V come back as zeros (OOB) ----
    if (wave == 0) {
        unsigned rem8 = (unsigned)(V - vbase) * (DDIM / 2);   // valid 8B elems
        tdm_load_to_lds(emb + (size_t)vbase * DDIM, ldsbase, TV * DDIM / 2, rem8);
        __builtin_amdgcn_s_wait_tensorcnt(0);
    }
    if (tid < KCOMP) sCk[tid] = cAll[tid];
    __syncthreads();

    // ---- build A fragments (hi/lo) in registers: wave owns mtile=wave ----
    // 16-bit A 16x32: lane l row=l%16; elem e -> K = 32*ks + e + 16*(l>=16)
    v16bf Ah[8], Al[8];
    {
        const float* rowp = sRaw + (wave * 16 + lr) * DDIM;
#pragma unroll
        for (int ks = 0; ks < 8; ++ks) {
#pragma unroll
            for (int e = 0; e < 16; ++e) {
                int kk = 32 * ks + e + 2 * hi8;
                float f = rowp[kk];
                __bf16 h = (__bf16)f;
                Ah[ks][e] = h;
                Al[ks][e] = (__bf16)(f - (float)h);
            }
        }
    }
    __syncthreads();  // raw tile dead; buffers may be overwritten

    // ---- prologue prefetch: U_0 -> buffer 0 ----
    if (wave == 0)
        tdm_load_to_lds(Ufrag, ldsbase, 8192, 8192);

    // ---- pipelined k-loop: compute on buf[k&1], prefetch k+1 (Dmu at k=63) ----
    for (int k = 0; k < KCOMP; ++k) {
        const __bf16* sBh = (const __bf16*)(smem + (unsigned)(k & 1) * 65536u);
        const __bf16* sBl = sBh + 16384;

        if (wave == 0) {
            int kn = k + 1;
            const __bf16* src = (kn < KCOMP) ? (Ufrag + (size_t)kn * 32768) : DmuFrag;
            tdm_load_to_lds(src, ldsbase + (unsigned)(kn & 1) * 65536u, 8192, 8192);
            __builtin_amdgcn_s_wait_tensorcnt(1);   // buffer k ready
        }
        __syncthreads();   // publish buffer k to all waves

        float uv[4];
#pragma unroll
        for (int nt = 0; nt < 4; ++nt) uv[nt] = u0[k * RDIM + nt * 16 + lr];

        float part[8] = {0, 0, 0, 0, 0, 0, 0, 0};
#pragma unroll
        for (int nt = 0; nt < 4; ++nt) {
            v8f acc = {};
#pragma unroll
            for (int ks = 0; ks < 8; ++ks) {
                int boff = ((nt * 8 + ks) * 32 + lane) * 16;
                v16bf bh = *(const v16bf*)(sBh + boff);
                v16bf bl = *(const v16bf*)(sBl + boff);
                acc = __builtin_amdgcn_wmma_f32_16x16x32_bf16(false, Ah[ks], false, bh, (short)0, acc, false, false);
                acc = __builtin_amdgcn_wmma_f32_16x16x32_bf16(false, Al[ks], false, bh, (short)0, acc, false, false);
                acc = __builtin_amdgcn_wmma_f32_16x16x32_bf16(false, Ah[ks], false, bl, (short)0, acc, false, false);
            }
#pragma unroll
            for (int i = 0; i < 8; ++i) {
                float z = acc[i] - uv[nt];
                part[i] += z * z;
            }
        }
        // reduce over the 16 lanes sharing a row group (xor stays in half)
#pragma unroll
        for (int m = 1; m < 16; m <<= 1) {
#pragma unroll
            for (int i = 0; i < 8; ++i) part[i] += __shfl_xor(part[i], m, 32);
        }
        if (lr == 0) {
#pragma unroll
            for (int i = 0; i < 8; ++i)
                sS2[k * TV + wave * 16 + hi8 + i] = part[i];
        }
        __syncthreads();   // buffer k consumed; safe for TDM into it at k+2
    }

    // ---- G2 = E @ Dmu from buf0 (Dinv prefetch overlaps): sQ = -2*G2 ----
    if (wave == 0) {
        tdm_load_to_lds(DinvFrag, ldsbase + 65536u, 8192, 8192);  // -> buf1
        __builtin_amdgcn_s_wait_tensorcnt(1);   // Dmu (buf0) ready
    }
    __syncthreads();
    {
        const __bf16* sBh = (const __bf16*)smem;          // buf0
        const __bf16* sBl = sBh + 16384;
#pragma unroll
        for (int nt = 0; nt < 4; ++nt) {
            v8f acc = {};
#pragma unroll
            for (int ks = 0; ks < 8; ++ks) {
                int boff = ((nt * 8 + ks) * 32 + lane) * 16;
                v16bf bh = *(const v16bf*)(sBh + boff);
                v16bf bl = *(const v16bf*)(sBl + boff);
                acc = __builtin_amdgcn_wmma_f32_16x16x32_bf16(false, Ah[ks], false, bh, (short)0, acc, false, false);
                acc = __builtin_amdgcn_wmma_f32_16x16x32_bf16(false, Al[ks], false, bh, (short)0, acc, false, false);
                acc = __builtin_amdgcn_wmma_f32_16x16x32_bf16(false, Ah[ks], false, bl, (short)0, acc, false, false);
            }
#pragma unroll
            for (int i = 0; i < 8; ++i)
                sQ[(nt * 16 + lr) * TV + wave * 16 + hi8 + i] = -2.0f * acc[i];
        }
    }

    // ---- square A fragments in registers: E -> E^2 (position-preserving) ----
#pragma unroll
    for (int ks = 0; ks < 8; ++ks) {
#pragma unroll
        for (int e = 0; e < 16; ++e) {
            float f = (float)Ah[ks][e] + (float)Al[ks][e];
            float q = f * f;
            __bf16 h = (__bf16)q;
            Ah[ks][e] = h;
            Al[ks][e] = (__bf16)(q - (float)h);
        }
    }

    if (wave == 0) __builtin_amdgcn_s_wait_tensorcnt(0);  // Dinv (buf1) ready
    __syncthreads();

    // ---- G1 = E^2 @ Dinv from buf1: sQ += G1 ----
    {
        const __bf16* sBh = (const __bf16*)(smem + 65536u); // buf1
        const __bf16* sBl = sBh + 16384;
#pragma unroll
        for (int nt = 0; nt < 4; ++nt) {
            v8f acc = {};
#pragma unroll
            for (int ks = 0; ks < 8; ++ks) {
                int boff = ((nt * 8 + ks) * 32 + lane) * 16;
                v16bf bh = *(const v16bf*)(sBh + boff);
                v16bf bl = *(const v16bf*)(sBl + boff);
                acc = __builtin_amdgcn_wmma_f32_16x16x32_bf16(false, Ah[ks], false, bh, (short)0, acc, false, false);
                acc = __builtin_amdgcn_wmma_f32_16x16x32_bf16(false, Al[ks], false, bh, (short)0, acc, false, false);
                acc = __builtin_amdgcn_wmma_f32_16x16x32_bf16(false, Ah[ks], false, bl, (short)0, acc, false, false);
            }
#pragma unroll
            for (int i = 0; i < 8; ++i)
                sQ[(nt * 16 + lr) * TV + wave * 16 + hi8 + i] += acc[i];
        }
    }
    __syncthreads();

    // ---- combine: lb[k][v] = cAll_k - 0.5*(G1 - 2*G2) + 0.5*s2 ----
    for (int o = tid; o < KCOMP * TV; o += 256) {
        int k = o >> 7, row = o & 127;
        int v = vbase + row;
        if (v < V)
            lb[(size_t)k * V + v] = sCk[k] - 0.5f * sQ[o] + 0.5f * sS2[o];
    }
}

// ---------------------------------------------------------------------------
// Kernel 4: out[b][v] = m_v + log( sum_k T[b][k] * exp(lb[k][v] - m_v) )
// ---------------------------------------------------------------------------
__global__ void __launch_bounds__(256)
reduce_kernel(const float* __restrict__ lb,
              const float* __restrict__ T,
              float* __restrict__ out,
              int V)
{
    __shared__ float sT[BDIM * KCOMP];
    for (int i = threadIdx.x; i < BDIM * KCOMP; i += 256) sT[i] = T[i];
    __syncthreads();

    int v = blockIdx.x * 256 + threadIdx.x;
    if (v >= V) return;

    float m = -3.0e38f;
#pragma unroll
    for (int k = 0; k < KCOMP; ++k) m = fmaxf(m, lb[(size_t)k * V + v]);

    float s[BDIM];
#pragma unroll
    for (int b = 0; b < BDIM; ++b) s[b] = 0.0f;

    for (int k = 0; k < KCOMP; ++k) {
        float p = expf(lb[(size_t)k * V + v] - m);
#pragma unroll
        for (int b = 0; b < BDIM; ++b) s[b] += sT[b * KCOMP + k] * p;
    }
#pragma unroll
    for (int b = 0; b < BDIM; ++b) out[(size_t)b * V + v] = m + logf(s[b]);
}

// ---------------------------------------------------------------------------
extern "C" void kernel_launch(void* const* d_in, const int* in_sizes, int n_in,
                              void* d_out, int out_size, void* d_ws, size_t ws_size,
                              hipStream_t stream) {
    const float* theta = (const float*)d_in[0];   // (B,K)
    const float* emb   = (const float*)d_in[1];   // (V,d)
    const float* mus   = (const float*)d_in[2];   // (K,d)
    const float* L     = (const float*)d_in[3];   // (K,d,r)
    const float* logd  = (const float*)d_in[4];   // (K,d)
    float* out = (float*)d_out;

    const int V = in_sizes[1] / DDIM;

    // workspace layout (bytes, all 256-aligned)
    char* ws = (char*)d_ws;
    size_t off = 0;
    float*  lb       = (float*)(ws + off);  off += (size_t)KCOMP * V * 4;           // 12.8 MB
    __bf16* Ufrag    = (__bf16*)(ws + off); off += (size_t)KCOMP * 32768 * 2;       // 4 MB
    __bf16* DinvFrag = (__bf16*)(ws + off); off += 32768 * 2;
    __bf16* DmuFrag  = (__bf16*)(ws + off); off += 32768 * 2;
    float*  u0       = (float*)(ws + off);  off += KCOMP * RDIM * 4;
    float*  cAll     = (float*)(ws + off);  off += 256;
    float*  T        = (float*)(ws + off);  off += BDIM * KCOMP * 4;

    precompute_kernel<<<KCOMP, 256, 0, stream>>>(mus, L, logd, Ufrag, DinvFrag,
                                                 DmuFrag, u0, cAll);
    softmax_kernel<<<BDIM, KCOMP, 0, stream>>>(theta, T);
    main_kernel<<<(V + TV - 1) / TV, 256, 0, stream>>>(emb, Ufrag, DinvFrag,
                                                       DmuFrag, u0, cAll, lb, V);
    reduce_kernel<<<(V + 255) / 256, 256, 0, stream>>>(lb, T, out, V);
}
